// SharedNATransformerLayer_48292612276770
// MI455X (gfx1250) — matrix operand order repaired and checked by reading, not compiled, not run on hardware
//
#include <hip/hip_runtime.h>
#include <hip/hip_bf16.h>

typedef _Float16 half_t;
typedef _Float16 v16h __attribute__((ext_vector_type(16)));
typedef float    v8f  __attribute__((ext_vector_type(8)));

#define TOKENS 2048
#define DMODEL 256
#define DFF    1024
#define NHEAD  8
#define DHEAD  32

// ---------------------------------------------------------------------------
// WMMA fragment helpers (contiguous k = 16*half + s order, 16-bit A 16x32 / B 32x16)
// ---------------------------------------------------------------------------
union FragU { uint4 u[2]; v16h v; };

// A fragment: row-major A[M x K] f16. Lane l: M = l&15, half = l>>4.
// VGPR0-7 = K (half*16 + 0..15): two b128 loads.
__device__ __forceinline__ v16h load_frag_A(const half_t* __restrict__ A, int row,
                                            int K, int kc, int halfsel) {
    const half_t* base = A + (long)row * K + kc + halfsel * 16;
    FragU f;
    f.u[0] = *(const uint4*)(base);
    f.u[1] = *(const uint4*)(base + 8);
    return f.v;
}

// B fragment from packed Bp[kchunk][n][k'] (k' = 0..31 contiguous per column).
// Lane l: N = l&15, half = l>>4; VGPR v = K' (half*16 + 2v, 2v+1): one contiguous
// 32-byte run -> two b128 loads.
__device__ __forceinline__ v16h load_frag_B(const half_t* __restrict__ Bp, long baseHalves) {
    const uint4* p = (const uint4*)(Bp + baseHalves);
    FragU f;
    f.u[0] = p[0];
    f.u[1] = p[1];
    return f.v;
}

__device__ __forceinline__ v8f wmma_f16(v16h a, v16h b, v8f c) {
    return __builtin_amdgcn_wmma_f32_16x16x32_f16(false, a, false, b, (short)0, c,
                                                  false, false);
}

// ---------------------------------------------------------------------------
// Weight packing: W fp32 [K x N] row-major -> Bp f16 [(K/32) x N x 32]
// Bp[(k>>5)*N*32 + n*32 + (k&31)] = (f16)W[k*N + n]
// ---------------------------------------------------------------------------
__global__ void pack_w_kernel(const float* __restrict__ W, half_t* __restrict__ Bp,
                              int K, int N) {
    long i = (long)blockIdx.x * 256 + threadIdx.x;
    long total = (long)K * N;
    if (i >= total) return;
    long per_chunk = (long)N * 32;
    int chunk = (int)(i / per_chunk);
    long rem  = i - (long)chunk * per_chunk;
    int n  = (int)(rem >> 5);
    int kk = (int)(rem & 31);
    int k  = (chunk << 5) + kk;
    Bp[i] = (half_t)W[(long)k * N + n];
}

// ---------------------------------------------------------------------------
// RMSNorm over D=256 (one block per token, 256 threads), fp32 in -> f16 out
// ---------------------------------------------------------------------------
__global__ void rmsnorm_f16_kernel(const float* __restrict__ X, const float* __restrict__ w,
                                   half_t* __restrict__ Y) {
    __shared__ float red[8];
    long base = (long)blockIdx.x * DMODEL;
    float v = X[base + threadIdx.x];
    float ss = v * v;
#pragma unroll
    for (int off = 16; off; off >>= 1) ss += __shfl_xor(ss, off, 32);
    if ((threadIdx.x & 31) == 0) red[threadIdx.x >> 5] = ss;
    __syncthreads();
    float s = red[0] + red[1] + red[2] + red[3] + red[4] + red[5] + red[6] + red[7];
    float rs = rsqrtf(s * (1.0f / DMODEL) + 1e-6f);
    Y[base + threadIdx.x] = (half_t)(v * rs * w[threadIdx.x]);
}

// ---------------------------------------------------------------------------
// Generic WMMA GEMM: each wave owns a 16 x (16*NT) output tile (one shared A
// fragment, NT B fragments, NT WMMAs per k-step). 8 waves per block.
// C[M x N] = A_h[M x K] @ Bp (packed). Epilogue by MODE:
//   0: out_f32 = acc + bias[n]                  (QKV projections)
//   1: out_f32 = acc + bias[n] + res            (O-projection + residual)
//   2: out_f32 = acc + res                      (final MLP projection + residual)
//   3: out_f16 = silu(acc1) * acc2   (two B operands; fused SwiGLU, no bias)
// All control flow is wave-uniform -> EXEC all ones around v_wmma (ISA 7.12 req).
// Prefetch is unconditional: speculative prefetch past the end is silently
// dropped (ISA 7.3), avoiding an in-loop branch + WMMA hazard NOPs.
// ---------------------------------------------------------------------------
template <int MODE, int NT>
__global__ void gemm16_kernel(const half_t* __restrict__ A, const half_t* __restrict__ Bp,
                              const half_t* __restrict__ Bp2, const float* __restrict__ bias,
                              const float* __restrict__ res, float* __restrict__ outF,
                              half_t* __restrict__ outH, int M, int N, int K) {
    const int wave = threadIdx.x >> 5;
    const int lane = threadIdx.x & 31;
    const int tile = blockIdx.x * 8 + wave;
    const int ntn  = N / (16 * NT);
    const int tm   = (tile / ntn) << 4;
    const int tn   = (tile % ntn) * (16 * NT);
    if (tm >= M) return;                       // whole-wave uniform exit

    const int halfsel = lane >> 4;
    const int l15     = lane & 15;
    const int rowA    = tm + l15;

    v8f c[NT];
    v8f c2[NT];
#pragma unroll
    for (int j = 0; j < NT; ++j) { c[j] = (v8f){}; c2[j] = (v8f){}; }

    const int nk = K >> 5;
#pragma unroll 2
    for (int kc = 0; kc < nk; ++kc) {
        const v16h a = load_frag_A(A, rowA, K, kc << 5, halfsel);
#pragma unroll
        for (int j = 0; j < NT; ++j) {
            const long bbase =
                ((long)kc * N + tn + j * 16 + l15) * 32 + halfsel * 16;
            const v16h b = load_frag_B(Bp, bbase);
            c[j] = wmma_f16(a, b, c[j]);
            if (MODE == 3) {
                const v16h b2 = load_frag_B(Bp2, bbase);
                c2[j] = wmma_f16(a, b2, c2[j]);
            }
        }
        // speculative next-chunk prefetch (global_prefetch_b8), branchless
        __builtin_prefetch(A + (long)rowA * K + ((kc + 1) << 5), 0, 1);
        __builtin_prefetch(Bp + ((long)(kc + 1) * N + tn + l15) * 32, 0, 1);
    }

#pragma unroll
    for (int j = 0; j < NT; ++j) {
        const int colB = tn + j * 16 + l15;
        const float bval = (MODE == 0 || MODE == 1) ? bias[colB] : 0.0f;
#pragma unroll
        for (int r = 0; r < 8; ++r) {
            const int  m = tm + r + (halfsel << 3);
            const long o = (long)m * N + colB;
            const float acc = c[j][r];
            if (MODE == 0) {
                outF[o] = acc + bval;
            } else if (MODE == 1) {
                outF[o] = acc + bval + res[o];
            } else if (MODE == 2) {
                outF[o] = acc + res[o];
            } else {  // MODE == 3: fused SwiGLU
                const float sl = acc / (1.0f + __expf(-acc));
                outH[o] = (half_t)(sl * c2[j][r]);
            }
        }
    }
}

// ---------------------------------------------------------------------------
// NA3D attention: one block per token (T=8 causal K=5; H=W=16 centered K=5),
// 8 waves = 8 heads, 32 lanes = DH channels. Online softmax over 125 neighbors,
// dot products via wave32 xor-shuffle reduction. Output f16 (next GEMM's A).
// ---------------------------------------------------------------------------
__global__ void na3d_attn_kernel(const float* __restrict__ q, const float* __restrict__ k,
                                 const float* __restrict__ v, half_t* __restrict__ out) {
    const int tok  = blockIdx.x;
    const int head = threadIdx.x >> 5;
    const int lane = threadIdx.x & 31;

    const int t  = tok >> 8;          // T=8
    const int hh = (tok >> 4) & 15;   // H=16
    const int ww = tok & 15;          // W=16
    const int hs  = min(max(hh - 2, 0), 11);   // clip(h-2, 0, H-K)
    const int wsx = min(max(ww - 2, 0), 11);

    const long ch = (long)head * DHEAD + lane;
    const float qv = q[(long)tok * DMODEL + ch] * 0.17677669529663687f; // DH^-0.5

    float m = -3.0e38f, s = 0.0f, acc = 0.0f;

    for (int kt = 0; kt < 5; ++kt) {
        const int ti = t - 4 + kt;
        if (ti < 0) continue;                         // causal, wave-uniform
#pragma unroll
        for (int kh = 0; kh < 5; ++kh) {
#pragma unroll
            for (int kw = 0; kw < 5; ++kw) {
                const int  nidx = (((ti << 4) + hs + kh) << 4) + wsx + kw;
                const long base = (long)nidx * DMODEL + ch;
                float d = qv * k[base];
#pragma unroll
                for (int off = 16; off; off >>= 1) d += __shfl_xor(d, off, 32);
                const float nm = fmaxf(m, d);
                const float sc = __expf(m - nm);
                const float p  = __expf(d - nm);
                s   = s * sc + p;
                acc = acc * sc + p * v[base];
                m   = nm;
            }
        }
    }
    out[(long)tok * DMODEL + ch] = (half_t)(acc / s);
}

// ---------------------------------------------------------------------------
// Launch
// ---------------------------------------------------------------------------
extern "C" void kernel_launch(void* const* d_in, const int* in_sizes, int n_in,
                              void* d_out, int out_size, void* d_ws, size_t ws_size,
                              hipStream_t stream) {
    const float* x       = (const float*)d_in[0];
    const float* norm1_w = (const float*)d_in[1];
    const float* norm2_w = (const float*)d_in[2];
    const float* wq = (const float*)d_in[3];
    const float* bq = (const float*)d_in[4];
    const float* wk = (const float*)d_in[5];
    const float* bk = (const float*)d_in[6];
    const float* wv = (const float*)d_in[7];
    const float* bv = (const float*)d_in[8];
    const float* wo = (const float*)d_in[9];
    const float* bo = (const float*)d_in[10];
    const float* w1 = (const float*)d_in[11];
    const float* w2 = (const float*)d_in[12];
    const float* w3 = (const float*)d_in[13];
    float* out = (float*)d_out;

    // Workspace carve-up (all offsets 256B-aligned)
    char* ws = (char*)d_ws;
    size_t off = 0;
    auto take = [&](size_t bytes) { char* p = ws + off; off += (bytes + 255) & ~(size_t)255; return p; };
    half_t* wqp  = (half_t*)take((size_t)DMODEL * DMODEL * 2);
    half_t* wkp  = (half_t*)take((size_t)DMODEL * DMODEL * 2);
    half_t* wvp  = (half_t*)take((size_t)DMODEL * DMODEL * 2);
    half_t* wop  = (half_t*)take((size_t)DMODEL * DMODEL * 2);
    half_t* w1p  = (half_t*)take((size_t)DMODEL * DFF * 2);
    half_t* w2p  = (half_t*)take((size_t)DMODEL * DFF * 2);
    half_t* w3p  = (half_t*)take((size_t)DFF * DMODEL * 2);
    half_t* xn_h = (half_t*)take((size_t)TOKENS * DMODEL * 2);
    float*  qb   = (float*) take((size_t)TOKENS * DMODEL * 4);
    float*  kb   = (float*) take((size_t)TOKENS * DMODEL * 4);
    float*  vb   = (float*) take((size_t)TOKENS * DMODEL * 4);
    half_t* at_h = (half_t*)take((size_t)TOKENS * DMODEL * 2);
    float*  hb   = (float*) take((size_t)TOKENS * DMODEL * 4);
    half_t* y_h  = (half_t*)take((size_t)TOKENS * DMODEL * 2);
    half_t* g_h  = (half_t*)take((size_t)TOKENS * DFF * 2);

    const dim3 blk(256);

    // 1) pack all weights fp32 -> f16 k-chunk-major
    pack_w_kernel<<<(DMODEL * DMODEL + 255) / 256, blk, 0, stream>>>(wq, wqp, DMODEL, DMODEL);
    pack_w_kernel<<<(DMODEL * DMODEL + 255) / 256, blk, 0, stream>>>(wk, wkp, DMODEL, DMODEL);
    pack_w_kernel<<<(DMODEL * DMODEL + 255) / 256, blk, 0, stream>>>(wv, wvp, DMODEL, DMODEL);
    pack_w_kernel<<<(DMODEL * DMODEL + 255) / 256, blk, 0, stream>>>(wo, wop, DMODEL, DMODEL);
    pack_w_kernel<<<(DMODEL * DFF    + 255) / 256, blk, 0, stream>>>(w1, w1p, DMODEL, DFF);
    pack_w_kernel<<<(DMODEL * DFF    + 255) / 256, blk, 0, stream>>>(w2, w2p, DMODEL, DFF);
    pack_w_kernel<<<(DFF    * DMODEL + 255) / 256, blk, 0, stream>>>(w3, w3p, DFF, DMODEL);

    // 2) rmsnorm(x) -> f16
    rmsnorm_f16_kernel<<<TOKENS, blk, 0, stream>>>(x, norm1_w, xn_h);

    // 3) QKV projections (WMMA, bias epilogue). NT=2: wave tile 16x32.
    const int g_qkv = (TOKENS / 16) * (DMODEL / 32) / 8;   // 128 blocks
    gemm16_kernel<0, 2><<<g_qkv, blk, 0, stream>>>(xn_h, wqp, nullptr, bq, nullptr, qb, nullptr, TOKENS, DMODEL, DMODEL);
    gemm16_kernel<0, 2><<<g_qkv, blk, 0, stream>>>(xn_h, wkp, nullptr, bk, nullptr, kb, nullptr, TOKENS, DMODEL, DMODEL);
    gemm16_kernel<0, 2><<<g_qkv, blk, 0, stream>>>(xn_h, wvp, nullptr, bv, nullptr, vb, nullptr, TOKENS, DMODEL, DMODEL);

    // 4) neighborhood attention -> f16
    na3d_attn_kernel<<<TOKENS, blk, 0, stream>>>(qb, kb, vb, at_h);

    // 5) O-projection + bias + residual(x) -> h
    gemm16_kernel<1, 2><<<g_qkv, blk, 0, stream>>>(at_h, wop, nullptr, bo, x, hb, nullptr, TOKENS, DMODEL, DMODEL);

    // 6) rmsnorm(h) -> f16
    rmsnorm_f16_kernel<<<TOKENS, blk, 0, stream>>>(hb, norm2_w, y_h);

    // 7) fused SwiGLU: g = silu(y@w1) * (y@w2) -> f16 (NT=2, two B operands)
    const int g_glu = (TOKENS / 16) * (DFF / 32) / 8;      // 512 blocks
    gemm16_kernel<3, 2><<<g_glu, blk, 0, stream>>>(y_h, w1p, w2p, nullptr, nullptr, nullptr, g_h, TOKENS, DFF, DMODEL);

    // 8) out = h + g @ w3
    gemm16_kernel<2, 2><<<g_qkv, blk, 0, stream>>>(g_h, w3p, nullptr, nullptr, hb, out, nullptr, TOKENS, DMODEL, DFF);
}
